// DocumentGCN_81836306858614
// MI455X (gfx1250) — matrix operand order repaired and checked by reading, not compiled, not run on hardware
//
#include <hip/hip_runtime.h>
#include <hip/hip_bf16.h>

#define HIDDEN 64
#define CH 8   // k-tiles staged per LDS chunk (8 * 4KB = 32KB, double-buffered = 64KB)

typedef __attribute__((ext_vector_type(16))) __bf16        v16bf;
typedef __attribute__((ext_vector_type(8)))  float         v8f;
typedef __attribute__((ext_vector_type(4)))  float         v4f;
typedef __attribute__((ext_vector_type(4)))  unsigned int  v4u;
typedef __attribute__((ext_vector_type(4)))  int           v4i;

#if __has_builtin(__builtin_amdgcn_global_load_async_to_lds_b128)
#define HAVE_ASYNC_LDS 1
#else
#define HAVE_ASYNC_LDS 0
#endif

// fp32 -> bf16 bits, round to nearest even
__device__ inline unsigned short f2bf(float f) {
  unsigned int u = __builtin_bit_cast(unsigned int, f);
  unsigned int r = 0x7FFFu + ((u >> 16) & 1u);
  return (unsigned short)((u + r) >> 16);
}

union BfVec {
  unsigned short s[16];
  v4u            q[2];
  v16bf          v;
};

__device__ inline float atomAddF(float* p, float v) {
  return __hip_atomic_fetch_add(p, v, __ATOMIC_RELAXED, __HIP_MEMORY_SCOPE_AGENT);
}

// 16B global -> LDS, async if the toolchain exposes the gfx1250 builtin.
// Builtin signature (from clang diagnostic): (int4 AS1*, int4 AS3*, imm, imm).
__device__ inline void lds_copy16(v4u* dst, const v4u* src) {
#if HAVE_ASYNC_LDS
  __builtin_amdgcn_global_load_async_to_lds_b128(
      (__attribute__((address_space(1))) v4i*)src,
      (__attribute__((address_space(3))) v4i*)dst, 0, 0);
#else
  *dst = *src;
#endif
}

__device__ inline void lds_fill_wait() {
#if HAVE_ASYNC_LDS
#if __has_builtin(__builtin_amdgcn_s_wait_asynccnt)
  __builtin_amdgcn_s_wait_asynccnt(0);
#else
  asm volatile("s_wait_asynccnt 0" ::: "memory");
#endif
#endif
}

// ---------------- small utility kernels ----------------

__global__ void fill1_kernel(float* __restrict__ p, int n) {
  int i = blockIdx.x * blockDim.x + threadIdx.x;
  if (i < n) p[i] = 1.0f;
}

__global__ void zero_kernel(float* __restrict__ p, int n) {
  int i = blockIdx.x * blockDim.x + threadIdx.x;
  if (i < n) p[i] = 0.0f;
}

__global__ void deg_edges_kernel(const int* __restrict__ dst, float* __restrict__ deg,
                                 long long nE) {
  long long e = (long long)blockIdx.x * blockDim.x + threadIdx.x;
  if (e < nE) atomAddF(&deg[dst[e]], 1.0f);
}

__global__ void rsqrt_kernel(float* __restrict__ p, int n) {
  int i = blockIdx.x * blockDim.x + threadIdx.x;
  if (i < n) p[i] = rsqrtf(p[i]);
}

// Pack W[K x 64] into WMMA B-operand bf16 lane layout (zero-padded past K).
__global__ void packB_kernel(const float* __restrict__ W, unsigned short* __restrict__ out,
                             int K, int ktiles) {
  int t = blockIdx.x * blockDim.x + threadIdx.x;
  int total = ktiles * 4 * 32;
  if (t >= total) return;
  int lane = t & 31;
  int nt   = (t >> 5) & 3;
  int kt   = t >> 7;                       // t == ((kt*4+nt)*32 + lane)
  int col  = nt * 16 + (lane & 15);
  int kg   = lane >> 4;
  unsigned short* o = out + (size_t)t * 16;
#pragma unroll
  for (int e = 0; e < 16; ++e) {
    int k = kt * 32 + kg * 8 + (e < 8 ? e : e + 8);
    float v = (k < K) ? W[(size_t)k * HIDDEN + col] : 0.0f;
    o[e] = f2bf(v);
  }
}

// ---------------- WMMA GEMM: C[M x 64] = (relu?)A[M x K] * Bpack ----------------

__device__ inline void loadA8v(const float* __restrict__ p, float* out) {
  const v4f* q = reinterpret_cast<const v4f*>(p);
  v4f a = q[0], b = q[1];
  out[0] = a.x; out[1] = a.y; out[2] = a.z; out[3] = a.w;
  out[4] = b.x; out[5] = b.y; out[6] = b.z; out[7] = b.w;
}

template <bool RELU>
__device__ inline void cvtA(const float* f0, const float* f1, BfVec& av) {
#pragma unroll
  for (int j = 0; j < 8; ++j) {
    float a = f0[j], b = f1[j];
    if (RELU) { a = fmaxf(a, 0.f); b = fmaxf(b, 0.f); }
    av.s[j] = f2bf(a);
    av.s[8 + j] = f2bf(b);
  }
}

template <bool RELU>
__global__ __launch_bounds__(256) void gemm_bf16_wmma(
    const float* __restrict__ A, const v4u* __restrict__ Bpack,
    float* __restrict__ C, int M, int K) {
  extern __shared__ char smem_raw[];
  v4u* sB = reinterpret_cast<v4u*>(smem_raw);   // [2][CH*256] v4u = 64KB

  const int lane = threadIdx.x & 31;
  const int wave = blockIdx.x * (blockDim.x >> 5) + (threadIdx.x >> 5);
  const int rowTiles = (M + 15) >> 4;
  int tile = wave;
  if (tile >= rowTiles) tile -= rowTiles;       // wrap: keeps every wave fully active
  const int row0 = tile << 4;

  // A-operand lane mapping (16-bit A 16x32): m = lane&15, kg = lane>>4
  const int m  = lane & 15;
  const int kg = lane >> 4;
  int rA = row0 + m; if (rA >= M) rA = M - 1;
  const float* __restrict__ arow = A + (size_t)rA * K;

  const int kfull   = K >> 5;                   // full 32-k tiles
  const int nchunks = (kfull + CH - 1) / CH;

  const v8f vzero = {0.f, 0.f, 0.f, 0.f, 0.f, 0.f, 0.f, 0.f};
  v8f acc[4] = {vzero, vzero, vzero, vzero};

  // stage chunk 0 of packed B into LDS buffer 0
  {
    int tiles0 = kfull < CH ? kfull : CH;
    int nv = tiles0 * 256;
    for (int i = threadIdx.x; i < nv; i += blockDim.x)
      lds_copy16(sB + i, Bpack + i);
  }

  for (int c = 0; c < nchunks; ++c) {
    lds_fill_wait();
    __syncthreads();                            // chunk c resident; prev buffer free

    if (c + 1 < nchunks) {                      // issue async fill of next chunk
      int rem = kfull - (c + 1) * CH;
      int tiles = rem < CH ? rem : CH;
      int nv = tiles * 256;
      const v4u* gs = Bpack + (size_t)(c + 1) * CH * 256;
      v4u* ld = sB + ((c + 1) & 1) * (CH * 256);
      for (int i = threadIdx.x; i < nv; i += blockDim.x)
        lds_copy16(ld + i, gs + i);
    }

    const v4u* bb = sB + (c & 1) * (CH * 256);
    int rem = kfull - c * CH;
    int tiles = rem < CH ? rem : CH;
    for (int i = 0; i < tiles; ++i) {
      const int ka = (c * CH + i) * 32 + kg * 8;
      float f0[8], f1[8];
      loadA8v(arow + ka, f0);
      loadA8v(arow + ka + 16, f1);
      if (ka + 288 < K) __builtin_prefetch(arow + ka + 256, 0, 1);

      BfVec av;
      cvtA<RELU>(f0, f1, av);

      const v4u* bt = bb + i * 256 + lane * 2;
      v4u breg[8];                              // hoist all 8 ds loads before WMMAs
#pragma unroll
      for (int nt = 0; nt < 4; ++nt) {
        breg[2 * nt]     = bt[nt * 64];
        breg[2 * nt + 1] = bt[nt * 64 + 1];
      }
#pragma unroll
      for (int nt = 0; nt < 4; ++nt) {
        BfVec bv;
        bv.q[0] = breg[2 * nt];
        bv.q[1] = breg[2 * nt + 1];
        acc[nt] = __builtin_amdgcn_wmma_f32_16x16x32_bf16(
            false, av.v, false, bv.v, (short)0, acc[nt], false, false);
      }
    }
  }

  // K tail (< 32): guarded elementwise A, B tile read once from global
  if (K & 31) {
    const int ka = kfull * 32 + kg * 8;
    float f0[8], f1[8];
#pragma unroll
    for (int j = 0; j < 8; ++j) {
      f0[j] = (ka + j      < K) ? arow[ka + j]      : 0.0f;
      f1[j] = (ka + 16 + j < K) ? arow[ka + 16 + j] : 0.0f;
    }
    BfVec av;
    cvtA<RELU>(f0, f1, av);
    const v4u* bt = Bpack + (size_t)kfull * 256 + lane * 2;
#pragma unroll
    for (int nt = 0; nt < 4; ++nt) {
      BfVec bv;
      bv.q[0] = bt[nt * 64];
      bv.q[1] = bt[nt * 64 + 1];
      acc[nt] = __builtin_amdgcn_wmma_f32_16x16x32_bf16(
          false, av.v, false, bv.v, (short)0, acc[nt], false, false);
    }
  }

  // D layout: col = lane&15; VGPR r holds row r (lanes 0-15) / r+8 (lanes 16-31)
  const int col0  = lane & 15;
  const int rbase = row0 + kg * 8;
#pragma unroll
  for (int nt = 0; nt < 4; ++nt) {
    float* cp = C + (size_t)rbase * HIDDEN + nt * 16 + col0;
#pragma unroll
    for (int r = 0; r < 8; ++r)
      if (rbase + r < M) cp[(size_t)r * HIDDEN] = acc[nt][r];
  }
}

// ---------------- GCN aggregation ----------------

__global__ void agg_init_kernel(const float* __restrict__ h, const float* __restrict__ dinv,
                                const float* __restrict__ bias, float* __restrict__ agg,
                                long long total) {
  long long t = (long long)blockIdx.x * blockDim.x + threadIdx.x;
  if (t >= total) return;
  int i = (int)(t >> 6), f = (int)(t & 63);
  float di = dinv[i];
  agg[t] = bias[f] + h[t] * di * di;
}

__global__ void agg_edges_kernel(const int* __restrict__ src, const int* __restrict__ dst,
                                 const float* __restrict__ h, const float* __restrict__ dinv,
                                 float* __restrict__ agg, long long nE) {
  long long t = (long long)blockIdx.x * blockDim.x + threadIdx.x;
  long long e = t >> 6;
  if (e >= nE) return;
  int f = (int)(t & 63);
  int s = src[e], d = dst[e];
  float w = dinv[s] * dinv[d];
  atomAddF(&agg[(size_t)d * HIDDEN + f], w * h[(size_t)s * HIDDEN + f]);
}

__global__ void pool_kernel(const float* __restrict__ agg, const int* __restrict__ batch,
                            float* __restrict__ sums, float* __restrict__ cnts,
                            long long total) {
  long long t = (long long)blockIdx.x * blockDim.x + threadIdx.x;
  if (t >= total) return;
  int i = (int)(t >> 6), f = (int)(t & 63);
  int g = batch[i];
  atomAddF(&sums[(size_t)g * HIDDEN + f], fmaxf(agg[t], 0.f));
  if (f == 0) atomAddF(&cnts[g], 1.0f);
}

__global__ void head_kernel(const float* __restrict__ sums, const float* __restrict__ cnts,
                            const float* __restrict__ Wc, const float* __restrict__ bc,
                            float* __restrict__ out, int NG, int NC) {
  int g = blockIdx.x * blockDim.x + threadIdx.x;
  if (g >= NG) return;
  float inv = 1.0f / fmaxf(cnts[g], 1.0f);
  float logits[32];
  for (int c = 0; c < NC; ++c) {
    float a = bc[c];
    for (int f = 0; f < HIDDEN; ++f) a += sums[(size_t)g * HIDDEN + f] * inv * Wc[f * NC + c];
    logits[c] = a;
  }
  float mx = logits[0];
  for (int c = 1; c < NC; ++c) mx = fmaxf(mx, logits[c]);
  float se = 0.f;
  for (int c = 0; c < NC; ++c) se += __expf(logits[c] - mx);
  float lse = mx + __logf(se);
  for (int c = 0; c < NC; ++c) out[(size_t)g * NC + c] = logits[c] - lse;
}

// ---------------- driver ----------------

static inline unsigned cdiv_ll(long long a, long long b) { return (unsigned)((a + b - 1) / b); }

extern "C" void kernel_launch(void* const* d_in, const int* in_sizes, int n_in,
                              void* d_out, int out_size, void* d_ws, size_t ws_size,
                              hipStream_t stream) {
  const float* x  = (const float*)d_in[0];
  const float* W1 = (const float*)d_in[1];
  const float* b1 = (const float*)d_in[2];
  const float* W2 = (const float*)d_in[3];
  const float* b2 = (const float*)d_in[4];
  const float* Wc = (const float*)d_in[5];
  const float* bc = (const float*)d_in[6];
  const int* ei   = (const int*)d_in[7];
  const int* batch = (const int*)d_in[8];
  float* out = (float*)d_out;

  const int       N  = in_sizes[8];
  const long long E  = (long long)in_sizes[7] / 2;
  const int       K1 = in_sizes[1] / HIDDEN;
  const int       NC = in_sizes[6];
  const int       NG = out_size / NC;

  const int* src = ei;
  const int* dst = ei + E;

  char* ws = (char*)d_ws;
  size_t off = 0;
  auto carve = [&](size_t bytes) -> void* {
    void* p = ws + off;
    off += (bytes + 255) & ~(size_t)255;
    return p;
  };
  float* dinv = (float*)carve((size_t)N * 4);
  float* bufA = (float*)carve((size_t)N * HIDDEN * 4);   // h1, later h2
  float* bufB = (float*)carve((size_t)N * HIDDEN * 4);   // agg1, later agg2
  const int kt1 = (K1 + 31) / 32;
  const int kt2 = (HIDDEN + 31) / 32;
  unsigned short* Wp1 = (unsigned short*)carve((size_t)kt1 * 4 * 32 * 16 * 2);
  unsigned short* Wp2 = (unsigned short*)carve((size_t)kt2 * 4 * 32 * 16 * 2);
  float* sums = (float*)carve((size_t)NG * HIDDEN * 4);
  float* cnts = (float*)carve((size_t)NG * 4);
  (void)ws_size; (void)n_in;

  const int TPB = 256;
  const long long NF = (long long)N * HIDDEN;
  const size_t LDSB = 2u * CH * 256u * 16u;              // 64 KB double buffer

  fill1_kernel<<<cdiv_ll(N, TPB), TPB, 0, stream>>>(dinv, N);
  deg_edges_kernel<<<cdiv_ll(E, TPB), TPB, 0, stream>>>(dst, dinv, E);
  rsqrt_kernel<<<cdiv_ll(N, TPB), TPB, 0, stream>>>(dinv, N);

  packB_kernel<<<cdiv_ll((long long)kt1 * 128, TPB), TPB, 0, stream>>>(W1, Wp1, K1, kt1);
  packB_kernel<<<cdiv_ll((long long)kt2 * 128, TPB), TPB, 0, stream>>>(W2, Wp2, HIDDEN, kt2);

  const int rowTiles   = (N + 15) / 16;
  const int gemmBlocks = (rowTiles + 7) / 8;             // 8 waves / block

  gemm_bf16_wmma<false><<<gemmBlocks, TPB, LDSB, stream>>>(x, (const v4u*)Wp1, bufA, N, K1);
  agg_init_kernel<<<cdiv_ll(NF, TPB), TPB, 0, stream>>>(bufA, dinv, b1, bufB, NF);
  agg_edges_kernel<<<cdiv_ll(E * HIDDEN, TPB), TPB, 0, stream>>>(src, dst, bufA, dinv, bufB, E);

  gemm_bf16_wmma<true><<<gemmBlocks, TPB, LDSB, stream>>>(bufB, (const v4u*)Wp2, bufA, N, HIDDEN);
  agg_init_kernel<<<cdiv_ll(NF, TPB), TPB, 0, stream>>>(bufA, dinv, b2, bufB, NF);
  agg_edges_kernel<<<cdiv_ll(E * HIDDEN, TPB), TPB, 0, stream>>>(src, dst, bufA, dinv, bufB, E);

  zero_kernel<<<cdiv_ll((long long)NG * HIDDEN, TPB), TPB, 0, stream>>>(sums, NG * HIDDEN);
  zero_kernel<<<1, TPB, 0, stream>>>(cnts, NG);
  pool_kernel<<<cdiv_ll(NF, TPB), TPB, 0, stream>>>(bufB, batch, sums, cnts, NF);
  head_kernel<<<1, 128, 0, stream>>>(sums, cnts, Wc, bc, out, NG, NC);
}
